// GIN_ESM_Attention_20907900797457
// MI455X (gfx1250) — matrix operand order repaired and checked by reading, not compiled, not run on hardware
//
#include <hip/hip_runtime.h>
#include <hip/hip_bf16.h>
#include <math.h>

typedef __attribute__((ext_vector_type(16))) _Float16 v16h;
typedef __attribute__((ext_vector_type(8)))  float    v8f;

#define HD 256

// ---------- helpers ----------
static __device__ __forceinline__ void atomicMaxF(float* addr, float v) {
  // IEEE trick: non-negative floats compare as signed ints; negative floats
  // compare reversed as unsigned ints.
  if (v >= 0.f) atomicMax((int*)addr, __float_as_int(v));
  else          atomicMin((unsigned int*)addr, __float_as_uint(v));
}

// A fragment: lane(half,m): e=0..15 -> K=half*16+e
static __device__ __forceinline__ v16h load_afrag(const float* __restrict__ a0) {
  v16h af;
#pragma unroll
  for (int e = 0; e < 8; ++e) {
    af[e]     = (_Float16)a0[e];
    af[8 + e] = (_Float16)a0[8 + e];
  }
  return af;
}
// B fragment: lane(half,n): e=0..15 -> K=half*16+e (contiguous in transposed W)
static __device__ __forceinline__ v16h load_bfrag(const _Float16* __restrict__ b0) {
  v16h bf;
#pragma unroll
  for (int e = 0; e < 16; ++e) bf[e] = b0[e];
  return bf;
}

// ---------- tiny utility kernels ----------
__global__ void k_fill(float* __restrict__ p, float v, int n) {
  int i = blockIdx.x * blockDim.x + threadIdx.x;
  if (i < n) p[i] = v;
}

// copy rows with stride change, zero-padding columns [ss, ds)
__global__ void k_copy_pad(const float* __restrict__ src, int ss,
                           float* __restrict__ dst, int ds, int rows) {
  int i = blockIdx.x * blockDim.x + threadIdx.x;
  int total = rows * ds;
  if (i >= total) return;
  int r = i / ds, c = i - r * ds;
  dst[i] = (c < ss) ? src[(size_t)r * ss + c] : 0.f;
}

// acc[dst[e], c] += h[src[e], c]
__global__ void k_scatter_add(const long long* __restrict__ srcI,
                              const long long* __restrict__ dstI,
                              const float* __restrict__ hsrc, int hs,
                              float* __restrict__ acc, int as, int C, int E) {
  int i = blockIdx.x * blockDim.x + threadIdx.x;
  int total = E * C;
  if (i >= total) return;
  int e = i / C, c = i - e * C;
  long long s = srcI[e], d = dstI[e];
  atomicAdd(acc + (size_t)d * as + c, hsrc[(size_t)s * hs + c]);
}

// f32 [K, Ho] -> f16 transposed [Ho, Kp], zero-padded K..Kp
__global__ void k_convT(const float* __restrict__ W, _Float16* __restrict__ Wt,
                        int K, int Kp, int Ho) {
  int i = blockIdx.x * blockDim.x + threadIdx.x;
  if (i >= Ho * Kp) return;
  int c = i / Kp, k = i - c * Kp;
  Wt[i] = (k < K) ? (_Float16)W[(size_t)k * Ho + c] : (_Float16)0.f;
}

// ---------- WMMA GEMM: C[rows,Ho] = act(A[rows,K] @ W[K,Ho] + bias) ----------
// One wave per 16x64 output strip (4 accumulators), software-pipelined over K:
// fragments for K-step kk+32 are loaded while the 4 WMMAs of step kk execute,
// so the loadcnt wait for them lands one WMMA-group later (double-buffered
// live ranges the allocator cannot collapse). W pre-converted f16, transposed
// [Ho][Kp] so each lane's B fragment is 16 contiguous halves (b128 pair).
// Optional row gather via rowidx. Requires Ho % 64 == 0, K % 32 == 0.
__global__ void k_gemm_wmma(const float* __restrict__ A, int lda,
                            const long long* __restrict__ rowidx,
                            const _Float16* __restrict__ Wt, int Kp,
                            const float* __restrict__ bias,
                            float* __restrict__ C, int ldc,
                            int rows, int K, int Ho, int relu) {
  int wid  = blockIdx.x * (blockDim.x >> 5) + (threadIdx.x >> 5);
  int lane = threadIdx.x & 31;
  int tilesN = Ho >> 6;                     // 64-wide strips
  int tiles  = (rows >> 4) * tilesN;
  if (wid >= tiles) return;                 // uniform per wave: EXEC stays all-1s
  int tm = wid / tilesN, tn = wid - tm * tilesN;
  int half = lane >> 4, idx = lane & 15;

  long long ar = tm * 16 + idx;
  if (rowidx) ar = rowidx[ar];              // gather mode
  const float* ap = A + ar * (long long)lda + half * 16;

  const _Float16* bp[4];
#pragma unroll
  for (int j = 0; j < 4; ++j)
    bp[j] = Wt + (size_t)(tn * 64 + j * 16 + idx) * Kp + half * 16;

  // ---- pipeline prologue: K-step 0 ----
  v8f acc[4] = {};
  v16h af = load_afrag(ap);
  v16h bf[4];
#pragma unroll
  for (int j = 0; j < 4; ++j) bf[j] = load_bfrag(bp[j]);

  // ---- steady state: load kk while computing kk-32 ----
  for (int kk = 32; kk < K; kk += 32) {
    v16h afn = load_afrag(ap + kk);
    v16h bfn[4];
#pragma unroll
    for (int j = 0; j < 4; ++j) bfn[j] = load_bfrag(bp[j] + kk);
#pragma unroll
    for (int j = 0; j < 4; ++j) {
      acc[j] = __builtin_amdgcn_wmma_f32_16x16x32_f16(
          false, af, false, bf[j], (short)0, acc[j], false, false);
    }
    af = afn;
#pragma unroll
    for (int j = 0; j < 4; ++j) bf[j] = bfn[j];
  }
  // ---- epilogue: last K-step ----
#pragma unroll
  for (int j = 0; j < 4; ++j) {
    acc[j] = __builtin_amdgcn_wmma_f32_16x16x32_f16(
        false, af, false, bf[j], (short)0, acc[j], false, false);
  }

#pragma unroll
  for (int j = 0; j < 4; ++j) {
    int col = tn * 64 + j * 16 + idx;
    float bc = bias ? bias[col] : 0.f;
#pragma unroll
    for (int v = 0; v < 8; ++v) {           // C/D: VGPR v -> row half*8+v
      int m = tm * 16 + half * 8 + v;
      float val = acc[j][v] + bc;
      if (relu) val = fmaxf(val, 0.f);
      C[(size_t)m * ldc + col] = val;
    }
  }
}

// ---------- BatchNorm over N rows (per-channel) ----------
__global__ void k_bn_stats(const float* __restrict__ A, float* __restrict__ sum,
                           float* __restrict__ sumsq, int rows) {
  int c = threadIdx.x;                      // 256 channels == blockDim
  int r0 = blockIdx.x * 256;
  int rend = r0 + 256; if (rend > rows) rend = rows;
  float s = 0.f, q = 0.f;
  for (int r = r0; r < rend; ++r) {
    float v = A[(size_t)r * HD + c];
    s += v; q += v * v;
  }
  atomicAdd(sum + c, s);
  atomicAdd(sumsq + c, q);
}

__global__ void k_bn_apply(const float* __restrict__ A, const float* __restrict__ sum,
                           const float* __restrict__ sumsq, const float* __restrict__ g,
                           const float* __restrict__ b, float* __restrict__ Hb,
                           int addres, int rows) {
  int i = blockIdx.x * blockDim.x + threadIdx.x;
  if (i >= rows * HD) return;
  int c = i & (HD - 1);
  float invN = 1.f / (float)rows;
  float mu  = sum[c] * invN;
  float var = sumsq[c] * invN - mu * mu;
  float v = (A[i] - mu) * rsqrtf(var + 1e-5f) * g[c] + b[c];
  v = fmaxf(v, 0.f);
  if (addres) Hb[i] += v; else Hb[i] = v;
}

// ---------- row LayerNorm (+opt ReLU), block(256)=row ----------
__global__ void k_ln(const float* __restrict__ X, const float* __restrict__ g,
                     const float* __restrict__ b, float* __restrict__ Y, int dorelu) {
  __shared__ float sm[256];
  int r = blockIdx.x, t = threadIdx.x;
  float v = X[(size_t)r * HD + t];
  sm[t] = v; __syncthreads();
  for (int s = 128; s > 0; s >>= 1) { if (t < s) sm[t] += sm[t + s]; __syncthreads(); }
  float mu = sm[0] * (1.f / HD);
  __syncthreads();
  float d = v - mu;
  sm[t] = d * d; __syncthreads();
  for (int s = 128; s > 0; s >>= 1) { if (t < s) sm[t] += sm[t + s]; __syncthreads(); }
  float y = d * rsqrtf(sm[0] * (1.f / HD) + 1e-5f) * g[t] + b[t];
  if (dorelu) y = fmaxf(y, 0.f);
  Y[(size_t)r * HD + t] = y;
}

// ---------- dots = sum(Q*K,-1)*SCALE, atomic segment max ----------
__global__ void k_rowdot(const float* __restrict__ Q, const float* __restrict__ Kc,
                         const long long* __restrict__ batch,
                         float* __restrict__ dots, float* __restrict__ m) {
  __shared__ float sm[256];
  int r = blockIdx.x, t = threadIdx.x;
  sm[t] = Q[(size_t)r * HD + t] * Kc[(size_t)r * HD + t];
  __syncthreads();
  for (int s = 128; s > 0; s >>= 1) { if (t < s) sm[t] += sm[t + s]; __syncthreads(); }
  if (t == 0) {
    float d = sm[0] * 0.125f;               // SCALE = (256/4)^-0.5
    dots[r] = d;
    atomicMaxF(m + batch[r], d);
  }
}

__global__ void k_expseg(const float* __restrict__ dots, const float* __restrict__ m,
                         const long long* __restrict__ batch,
                         float* __restrict__ e, float* __restrict__ s, int rows) {
  int r = blockIdx.x * blockDim.x + threadIdx.x;
  if (r >= rows) return;
  long long b = batch[r];
  float ev = expf(dots[r] - m[b]);
  e[r] = ev;
  atomicAdd(s + b, ev);
}

// ---------- out = w*V + h; LN; segment-sum into drug_vec (fused) ----------
__global__ void k_attend(const float* __restrict__ V, const float* __restrict__ Hb,
                         const float* __restrict__ e, const float* __restrict__ s,
                         const long long* __restrict__ batch,
                         const float* __restrict__ g, const float* __restrict__ b,
                         float* __restrict__ drug) {
  __shared__ float sm[256];
  int r = blockIdx.x, t = threadIdx.x;
  long long bt = batch[r];
  float w = e[r] / (s[bt] + 1e-16f);
  float v = w * V[(size_t)r * HD + t] + Hb[(size_t)r * HD + t];
  sm[t] = v; __syncthreads();
  for (int st = 128; st > 0; st >>= 1) { if (t < st) sm[t] += sm[t + st]; __syncthreads(); }
  float mu = sm[0] * (1.f / HD);
  __syncthreads();
  float d = v - mu;
  sm[t] = d * d; __syncthreads();
  for (int st = 128; st > 0; st >>= 1) { if (t < st) sm[t] += sm[t + st]; __syncthreads(); }
  float y = d * rsqrtf(sm[0] * (1.f / HD) + 1e-5f) * g[t] + b[t];
  atomicAdd(drug + (size_t)bt * HD + t, y);
}

__global__ void k_concat(const float* __restrict__ drug, const float* __restrict__ pv,
                         float* __restrict__ cat, int B) {
  int i = blockIdx.x * blockDim.x + threadIdx.x;
  if (i >= B * 512) return;
  int r = i >> 9, c = i & 511;
  cat[i] = (c < HD) ? drug[(size_t)r * HD + c] : pv[(size_t)r * HD + (c - HD)];
}

// ---------- final [B,256] @ [256,1] + b ----------
__global__ void k_final(const float* __restrict__ Z, const float* __restrict__ w3,
                        const float* __restrict__ b3, float* __restrict__ out) {
  __shared__ float sm[256];
  int r = blockIdx.x, t = threadIdx.x;
  sm[t] = Z[(size_t)r * HD + t] * w3[t];
  __syncthreads();
  for (int s = 128; s > 0; s >>= 1) { if (t < s) sm[t] += sm[t + s]; __syncthreads(); }
  if (t == 0) out[r] = sm[0] + b3[0];
}

// ================= host =================
extern "C" void kernel_launch(void* const* d_in, const int* in_sizes, int n_in,
                              void* d_out, int out_size, void* d_ws, size_t ws_size,
                              hipStream_t stream) {
  (void)in_sizes; (void)n_in; (void)out_size; (void)ws_size;
  const int N = 102400, E = 409600, B = 2048;
  const int F_ATOM = 70, KP0 = 96, F_PROT = 480;

  const float*     x     = (const float*)d_in[0];
  const long long* eidx  = (const long long*)d_in[1];
  const long long* esrc  = eidx;
  const long long* edst  = eidx + E;
  const long long* batch = (const long long*)d_in[2];
  const float*     prot  = (const float*)d_in[3];

  const float *ginw1[4], *ginb1[4], *ginw2[4], *ginb2[4], *bng[4], *bnb[4];
  for (int i = 0; i < 4; ++i) {
    int base = 4 + 6 * i;
    ginw1[i] = (const float*)d_in[base + 0];
    ginb1[i] = (const float*)d_in[base + 1];
    ginw2[i] = (const float*)d_in[base + 2];
    ginb2[i] = (const float*)d_in[base + 3];
    bng[i]   = (const float*)d_in[base + 4];
    bnb[i]   = (const float*)d_in[base + 5];
  }
  const float* proj_w = (const float*)d_in[28]; const float* proj_b = (const float*)d_in[29];
  const float* pln_g  = (const float*)d_in[30]; const float* pln_b  = (const float*)d_in[31];
  const float* q_w = (const float*)d_in[32];    const float* q_b = (const float*)d_in[33];
  const float* k_w = (const float*)d_in[34];    const float* k_b = (const float*)d_in[35];
  const float* v_w = (const float*)d_in[36];    const float* v_b = (const float*)d_in[37];
  const float* att_g = (const float*)d_in[38];  const float* att_b = (const float*)d_in[39];
  const float* p1w = (const float*)d_in[40];    const float* p1b = (const float*)d_in[41];
  const float* p2w = (const float*)d_in[42];    const float* p2b = (const float*)d_in[43];
  const float* p3w = (const float*)d_in[44];    const float* p3b = (const float*)d_in[45];

  // ---- workspace carve ----
  char* wsp = (char*)d_ws;
  size_t off = 0;
  auto take = [&](size_t bytes) -> void* {
    void* p = wsp + off; off += (bytes + 255) & ~(size_t)255; return p;
  };
  float* bufA = (float*)take((size_t)N * HD * 4 + 4096);
  float* bufB = (float*)take((size_t)N * HD * 4 + 4096);
  float* bufH = (float*)take((size_t)N * HD * 4 + 4096);
  _Float16* wh_g0w1 = (_Float16*)take((size_t)KP0 * HD * 2);
  _Float16* whW1[4]; whW1[0] = wh_g0w1;
  _Float16* whW2[4];
  whW2[0] = (_Float16*)take((size_t)HD * HD * 2);
  for (int i = 1; i < 4; ++i) {
    whW1[i] = (_Float16*)take((size_t)HD * HD * 2);
    whW2[i] = (_Float16*)take((size_t)HD * HD * 2);
  }
  _Float16* wh_proj = (_Float16*)take((size_t)F_PROT * HD * 2);
  _Float16* wh_q = (_Float16*)take((size_t)HD * HD * 2);
  _Float16* wh_k = (_Float16*)take((size_t)HD * HD * 2);
  _Float16* wh_v = (_Float16*)take((size_t)HD * HD * 2);
  _Float16* wh_p1 = (_Float16*)take((size_t)512 * 512 * 2);
  _Float16* wh_p2 = (_Float16*)take((size_t)512 * HD * 2);
  float* bnsum = (float*)take(512 * 4);
  float* pvb   = (float*)take((size_t)B * HD * 4);
  float* dots  = (float*)take((size_t)N * 4);
  float* ebuf  = (float*)take((size_t)N * 4);
  float* mbuf  = (float*)take((size_t)B * 4);
  float* sbuf  = (float*)take((size_t)B * 4);
  float* drug  = (float*)take((size_t)B * HD * 4);
  float* catb  = (float*)take((size_t)B * 512 * 4);
  float* z1    = (float*)take((size_t)B * 512 * 4);
  float* z2    = (float*)take((size_t)B * HD * 4);

  auto convT = [&](const float* W, _Float16* Wt, int K, int Kp, int Ho) {
    int n = Ho * Kp;
    k_convT<<<(n + 255) / 256, 256, 0, stream>>>(W, Wt, K, Kp, Ho);
  };
  auto gemm = [&](const float* A, int lda, const long long* ridx, const _Float16* Wt,
                  int Kp, const float* bias, float* Cc, int ldc, int rows, int K,
                  int Ho, int relu) {
    int tiles = (rows / 16) * (Ho / 64);    // 16x64 strip per wave
    k_gemm_wmma<<<(tiles + 7) / 8, 256, 0, stream>>>(A, lda, ridx, Wt, Kp, bias, Cc,
                                                     ldc, rows, K, Ho, relu);
  };

  // ---- convert all weights to f16 (transposed, K padded) ----
  convT(ginw1[0], whW1[0], F_ATOM, KP0, HD);
  convT(ginw2[0], whW2[0], HD, HD, HD);
  for (int i = 1; i < 4; ++i) { convT(ginw1[i], whW1[i], HD, HD, HD); convT(ginw2[i], whW2[i], HD, HD, HD); }
  convT(proj_w, wh_proj, F_PROT, F_PROT, HD);
  convT(q_w, wh_q, HD, HD, HD); convT(k_w, wh_k, HD, HD, HD); convT(v_w, wh_v, HD, HD, HD);
  convT(p1w, wh_p1, 512, 512, 512);
  convT(p2w, wh_p2, 512, 512, HD);

  // ---- GIN layers ----
  for (int i = 0; i < 4; ++i) {
    const float* hin = (i == 0) ? x : bufH;
    int din  = (i == 0) ? F_ATOM : HD;
    int dpad = (i == 0) ? KP0 : HD;
    int tA = N * dpad;
    k_copy_pad<<<(tA + 255) / 256, 256, 0, stream>>>(hin, din, bufA, dpad, N);  // z := h (+pad)
    int tS = E * din;
    k_scatter_add<<<(tS + 255) / 256, 256, 0, stream>>>(esrc, edst, hin, din, bufA, dpad, din, E);
    gemm(bufA, dpad, nullptr, whW1[i], dpad, ginb1[i], bufB, HD, N, dpad, HD, 1);
    gemm(bufB, HD, nullptr, whW2[i], HD, ginb2[i], bufA, HD, N, HD, HD, 0);
    k_fill<<<2, 256, 0, stream>>>(bnsum, 0.f, 512);
    k_bn_stats<<<(N + 255) / 256, 256, 0, stream>>>(bufA, bnsum, bnsum + 256, N);
    k_bn_apply<<<((N * HD) + 255) / 256, 256, 0, stream>>>(bufA, bnsum, bnsum + 256,
                                                           bng[i], bnb[i], bufH, i > 0 ? 1 : 0, N);
  }

  // ---- protein projector: pv = relu(LN(prot @ proj_w + b)) ----
  gemm(prot, F_PROT, nullptr, wh_proj, F_PROT, proj_b, z1, HD, B, F_PROT, HD, 0);
  k_ln<<<B, 256, 0, stream>>>(z1, pln_g, pln_b, pvb, 1);

  // ---- cross attention ----
  gemm(pvb, HD, batch, wh_q, HD, q_b, bufA, HD, N, HD, HD, 0);      // Q = pv[batch] @ Wq
  gemm(bufH, HD, nullptr, wh_k, HD, k_b, bufB, HD, N, HD, HD, 0);   // K
  k_fill<<<(B + 255) / 256, 256, 0, stream>>>(mbuf, -INFINITY, B);
  k_fill<<<(B + 255) / 256, 256, 0, stream>>>(sbuf, 0.f, B);
  k_rowdot<<<N, 256, 0, stream>>>(bufA, bufB, batch, dots, mbuf);
  k_expseg<<<(N + 255) / 256, 256, 0, stream>>>(dots, mbuf, batch, ebuf, sbuf, N);
  gemm(bufH, HD, nullptr, wh_v, HD, v_b, bufA, HD, N, HD, HD, 0);   // V (overwrites Q)
  k_fill<<<((B * HD) + 255) / 256, 256, 0, stream>>>(drug, 0.f, B * HD);
  k_attend<<<N, 256, 0, stream>>>(bufA, bufH, ebuf, sbuf, batch, att_g, att_b, drug);

  // ---- predictor head ----
  k_concat<<<((B * 512) + 255) / 256, 256, 0, stream>>>(drug, pvb, catb, B);
  gemm(catb, 512, nullptr, wh_p1, 512, p1b, z1, 512, B, 512, 512, 1);
  gemm(z1, 512, nullptr, wh_p2, 512, p2b, z2, HD, B, 512, HD, 1);
  k_final<<<B, 256, 0, stream>>>(z2, p3w, p3b, (float*)d_out);
}
